// DeformableTransformerLayer_77094662963365
// MI455X (gfx1250) — matrix operand order repaired and checked by reading, not compiled, not run on hardware
//
#include <hip/hip_runtime.h>
#include <hip/hip_bf16.h>
#include <math.h>

// ---------------- problem constants (from reference) ----------------
#define EMBED   256
#define HEADS   8
#define HD      32      // EMBED / HEADS -- exactly one wave32 lane per channel
#define LEVELS  3
#define POINTS  4
#define NQ      40000
#define NV      52500   // 200*200 + 100*100 + 50*50
#define NOFF    192     // HEADS*LEVELS*POINTS*2
#define NATT    96      // HEADS*LEVELS*POINTS
#define FFN     1024

typedef unsigned short u16;
typedef __attribute__((ext_vector_type(16))) __bf16 bf16x16;
typedef __attribute__((ext_vector_type(8)))  float  f32x8;

// ---------------- bf16 helpers (RNE) ----------------
__device__ __forceinline__ u16 f2bf(float f) {
  unsigned int u = __float_as_uint(f);
  unsigned int r = u + 0x7FFFu + ((u >> 16) & 1u);
  return (u16)(r >> 16);
}
__device__ __forceinline__ float bf2f(u16 h) {
  return __uint_as_float(((unsigned int)h) << 16);
}

// ---------------- f32 -> bf16 conversion (row-major, for A operands) ----------------
__global__ __launch_bounds__(256) void cvt_bf16_kernel(const float* __restrict__ s,
                                                       u16* __restrict__ d, int n) {
  int i = blockIdx.x * 256 + threadIdx.x;
  if (i < n) d[i] = f2bf(s[i]);
}

// ---------------- f32 (K x N) -> bf16 transposed (N x K), for B operands --------------
// Weights are tiny (<=0.5MB) and L2-resident on a 192MB-L2 part; transposing them
// once lets every B WMMA fragment be two contiguous 16B global loads (no LDS staging).
__global__ __launch_bounds__(256) void cvt_bf16_t_kernel(const float* __restrict__ s,
                                                         u16* __restrict__ d,
                                                         int K, int N) {
  int i = blockIdx.x * 256 + threadIdx.x;
  if (i < K * N) {
    int k = i / N, n = i - k * N;
    d[(size_t)n * K + k] = f2bf(s[i]);
  }
}

// ---------------- LayerNorm (one block of 256 threads per row) ----------------
// out_f32 (optional): normalized row (pre-pos)   [identity for attention]
// out_bf : bf16(normalized (+ pos if pos != null))
__global__ __launch_bounds__(256)
void ln_kernel(const float* __restrict__ x, const float* __restrict__ pos,
               const float* __restrict__ g, const float* __restrict__ b,
               float* __restrict__ out_f32, u16* __restrict__ out_bf) {
  __shared__ float sh[8];
  const int row = blockIdx.x, t = threadIdx.x;
  const size_t base = (size_t)row * EMBED;
  float v = x[base + t];
  float s = v;
#pragma unroll
  for (int o = 16; o > 0; o >>= 1) s += __shfl_xor(s, o, 32);
  if ((t & 31) == 0) sh[t >> 5] = s;
  __syncthreads();
  float tot = 0.f;
#pragma unroll
  for (int i = 0; i < 8; ++i) tot += sh[i];
  const float mean = tot * (1.0f / EMBED);
  __syncthreads();
  const float d = v - mean;
  float s2 = d * d;
#pragma unroll
  for (int o = 16; o > 0; o >>= 1) s2 += __shfl_xor(s2, o, 32);
  if ((t & 31) == 0) sh[t >> 5] = s2;
  __syncthreads();
  float tot2 = 0.f;
#pragma unroll
  for (int i = 0; i < 8; ++i) tot2 += sh[i];
  const float var = tot2 * (1.0f / EMBED);
  float y = d * rsqrtf(var + 1e-5f) * g[t] + b[t];
  if (out_f32) out_f32[base + t] = y;
  if (pos) y += pos[base + t];
  out_bf[base + t] = f2bf(y);
}

// ---------------- bf16 WMMA GEMM: C[M,N] = A[M,K] @ BT[N,K]^T + bias (+adds) ----------
// Block: 256 threads = 8 waves, all tiled along M -> block tile 256(M) x 32(N).
// Each wave owns a 32x32 output tile: 2 A-fragments x 2 B-fragments -> 4
// v_wmma_f32_16x16x32_bf16 per K-step. B is pre-transposed (N-major), so every
// fragment (A and B) is two contiguous 16B global loads; no LDS, no barriers in
// the hot loop -- B streams from L2 (weights are L2-resident).
//
// ISA fragment layouts used:
//   A (16x32 bf16): lane<16 row M=lane, K = {kb..kb+8, kb+16..kb+24}, kb=kk
//                   lane>=16 same row, kb=kk+8
//   B (32x16 bf16): lane<16 col N=lane, K = {kb..kb+8, kb+16..kb+24}, kb=kk
//                   lane>=16 same col, kb=kk+8
#define EPI_F32       0
#define EPI_BF16      1
#define EPI_GELU_BF16 2

__global__ __launch_bounds__(256)
void gemm_bf16_wmma(const u16* __restrict__ A, const u16* __restrict__ BT,
                    const float* __restrict__ bias,
                    const float* __restrict__ add0, const float* __restrict__ add1,
                    void* __restrict__ Cout, int M, int N, int K, int epi) {
  const int wave = threadIdx.x >> 5;
  const int lane = threadIdx.x & 31;
  const int half = lane >> 4;              // 0: lanes 0-15, 1: lanes 16-31
  const int l16  = lane & 15;
  const int m0 = blockIdx.x * 256 + wave * 32;
  const int n0 = blockIdx.y * 32;

  int ar0 = m0 + l16;      if (ar0 >= M) ar0 = M - 1;   // clamp; stores predicated
  int ar1 = m0 + 16 + l16; if (ar1 >= M) ar1 = M - 1;
  const u16* Arow0 = A + (size_t)ar0 * K;
  const u16* Arow1 = A + (size_t)ar1 * K;
  const u16* Bcol0 = BT + (size_t)(n0 + l16) * K;        // column n0+l16
  const u16* Bcol1 = BT + (size_t)(n0 + 16 + l16) * K;   // column n0+16+l16

  f32x8 acc00 = {}, acc01 = {}, acc10 = {}, acc11 = {};

#pragma unroll 2
  for (int kk = 0; kk < K; kk += 32) {
    if (kk + 32 < K) {
      __builtin_prefetch(Arow0 + kk + 32, 0, 0);
      __builtin_prefetch(Arow1 + kk + 32, 0, 0);
    }
    const int kb = kk + half * 8;        // A per-lane chunk base
    const int kc = kk + half * 8;        // B per-lane chunk base

    union { uint4 u[2]; bf16x16 v; } fa0, fa1, fb0, fb1;
    fa0.u[0] = *(const uint4*)(Arow0 + kb);
    fa0.u[1] = *(const uint4*)(Arow0 + kb + 16);
    fa1.u[0] = *(const uint4*)(Arow1 + kb);
    fa1.u[1] = *(const uint4*)(Arow1 + kb + 16);
    fb0.u[0] = *(const uint4*)(Bcol0 + kc);
    fb0.u[1] = *(const uint4*)(Bcol0 + kc + 16);
    fb1.u[0] = *(const uint4*)(Bcol1 + kc);
    fb1.u[1] = *(const uint4*)(Bcol1 + kc + 16);

    acc00 = __builtin_amdgcn_wmma_f32_16x16x32_bf16(false, fa0.v, false, fb0.v,
                                                    (short)0, acc00, false, false);
    acc01 = __builtin_amdgcn_wmma_f32_16x16x32_bf16(false, fa0.v, false, fb1.v,
                                                    (short)0, acc01, false, false);
    acc10 = __builtin_amdgcn_wmma_f32_16x16x32_bf16(false, fa1.v, false, fb0.v,
                                                    (short)0, acc10, false, false);
    acc11 = __builtin_amdgcn_wmma_f32_16x16x32_bf16(false, fa1.v, false, fb1.v,
                                                    (short)0, acc11, false, false);
  }

  // epilogue: C/D layout -- lane holds column l16 (+16 for nt=1),
  // VGPR r holds row half*8 + r of each 16x16 tile; row tile rt adds 16.
#pragma unroll
  for (int rt = 0; rt < 2; ++rt) {
#pragma unroll
    for (int nt = 0; nt < 2; ++nt) {
      f32x8 acc = rt ? (nt ? acc11 : acc10) : (nt ? acc01 : acc00);
      const int col = n0 + nt * 16 + l16;
      const float bv = bias ? bias[col] : 0.0f;
#pragma unroll
      for (int r = 0; r < 8; ++r) {
        const int row = m0 + rt * 16 + half * 8 + r;
        if (row < M) {
          const size_t off = (size_t)row * N + col;
          float x = acc[r] + bv;
          if (add0) x += add0[off];
          if (add1) x += add1[off];
          if (epi == EPI_GELU_BF16) {
            x = 0.5f * x * (1.0f + erff(x * 0.70710678118654752f));   // exact gelu
            ((u16*)Cout)[off] = f2bf(x);
          } else if (epi == EPI_BF16) {
            ((u16*)Cout)[off] = f2bf(x);
          } else {
            ((float*)Cout)[off] = x;
          }
        }
      }
    }
  }
}

// ---------------- softmax over LEVELS*POINTS = 12 (in place) ----------------
__global__ __launch_bounds__(256) void softmax12_kernel(float* __restrict__ a, int nrows) {
  int r = blockIdx.x * 256 + threadIdx.x;
  if (r >= nrows) return;
  float* p = a + (size_t)r * 12;
  float v[12], m = -1e30f;
#pragma unroll
  for (int i = 0; i < 12; ++i) { v[i] = p[i]; m = fmaxf(m, v[i]); }
  float s = 0.f;
#pragma unroll
  for (int i = 0; i < 12; ++i) { v[i] = __expf(v[i] - m); s += v[i]; }
  const float inv = 1.0f / s;
#pragma unroll
  for (int i = 0; i < 12; ++i) p[i] = v[i] * inv;
}

// ---------------- deformable bilinear sampling ----------------
// One wave per (query, head); lane == head-dim channel (HD == 32 == wave32).
// Scalar weights are wave-uniform; per-corner value gathers are 64B coalesced.
__global__ __launch_bounds__(256)
void sample_kernel(const u16* __restrict__ vbf, const float* __restrict__ offs,
                   const float* __restrict__ aw, const float* __restrict__ ref,
                   u16* __restrict__ samp) {
  const int Hs[3] = {200, 100, 50};
  const int Ws[3] = {200, 100, 50};
  const int St[3] = {0, 40000, 50000};
  const int wave = threadIdx.x >> 5;
  const int lane = threadIdx.x & 31;
  const int gid  = blockIdx.x * 8 + wave;      // q*HEADS + head
  const int q = gid >> 3, head = gid & 7;
  const float* offq = offs + (size_t)q * NOFF + head * (LEVELS * POINTS * 2);
  const float* awq  = aw   + (size_t)q * NATT + head * (LEVELS * POINTS);
  const float* refq = ref  + (size_t)q * (LEVELS * 2);
  float acc = 0.f;
#pragma unroll
  for (int l = 0; l < LEVELS; ++l) {
    const int H = Hs[l], W = Ws[l], st = St[l];
    const float rx = refq[l * 2 + 0], ry = refq[l * 2 + 1];
#pragma unroll
    for (int p = 0; p < POINTS; ++p) {
      const float ox = offq[l * 8 + p * 2 + 0];
      const float oy = offq[l * 8 + p * 2 + 1];
      const float a  = awq[l * 4 + p];
      const float x = (rx + ox / (float)W) * (float)W - 0.5f;
      const float y = (ry + oy / (float)H) * (float)H - 0.5f;
      const float x0f = floorf(x), y0f = floorf(y);
      const float tx = x - x0f, ty = y - y0f;
      const int x0 = (int)x0f, y0 = (int)y0f;
#pragma unroll
      for (int dy = 0; dy < 2; ++dy) {
#pragma unroll
        for (int dx = 0; dx < 2; ++dx) {
          const int xi = x0 + dx, yi = y0 + dy;
          const bool valid = (xi >= 0) & (xi < W) & (yi >= 0) & (yi < H);
          if (valid) {  // wave-uniform branch
            const float wgt = (dx ? tx : 1.f - tx) * (dy ? ty : 1.f - ty) * a;
            const size_t idx = (size_t)(st + yi * W + xi) * EMBED + head * HD + lane;
            acc += wgt * bf2f(vbf[idx]);
          }
        }
      }
    }
  }
  samp[(size_t)q * EMBED + head * HD + lane] = f2bf(acc);
}

// ---------------- host orchestration ----------------
extern "C" void kernel_launch(void* const* d_in, const int* in_sizes, int n_in,
                              void* d_out, int out_size, void* d_ws, size_t ws_size,
                              hipStream_t stream) {
  (void)in_sizes; (void)n_in; (void)out_size; (void)ws_size;
  const float* query = (const float*)d_in[0];
  const float* value = (const float*)d_in[1];
  const float* qpos  = (const float*)d_in[2];
  const float* refp  = (const float*)d_in[3];
  // d_in[4] spatial_shapes, d_in[5] level_start_index: compile-time constants
  const float* n1g  = (const float*)d_in[6];
  const float* n1b  = (const float*)d_in[7];
  const float* Woff = (const float*)d_in[8];
  const float* boff = (const float*)d_in[9];
  const float* Watt = (const float*)d_in[10];
  const float* batt = (const float*)d_in[11];
  const float* Wval = (const float*)d_in[12];
  const float* bval = (const float*)d_in[13];
  const float* Wout = (const float*)d_in[14];
  const float* bout = (const float*)d_in[15];
  const float* n2g  = (const float*)d_in[16];
  const float* n2b  = (const float*)d_in[17];
  const float* W1   = (const float*)d_in[18];
  const float* b1   = (const float*)d_in[19];
  const float* W2   = (const float*)d_in[20];
  const float* b2   = (const float*)d_in[21];

  // workspace carve (d_ws is 256B aligned from hipMalloc)
  char* ws = (char*)d_ws;
  size_t off = 0;
  auto carve = [&](size_t bytes) -> char* {
    char* p = ws + off;
    off += (bytes + 255) & ~(size_t)255;
    return p;
  };
  u16*   value_bf = (u16*)  carve((size_t)NV * EMBED * 2);
  u16*   qp_bf    = (u16*)  carve((size_t)NQ * EMBED * 2);
  float* q_id     = (float*)carve((size_t)NQ * EMBED * 4);
  float* offs_f   = (float*)carve((size_t)NQ * NOFF * 4);
  float* aw_f     = (float*)carve((size_t)NQ * NATT * 4);
  u16*   v_bf     = (u16*)  carve((size_t)NV * EMBED * 2);
  u16*   samp_bf  = (u16*)  carve((size_t)NQ * EMBED * 2);
  float* x_f      = (float*)carve((size_t)NQ * EMBED * 4);
  u16*   h_bf     = (u16*)  carve((size_t)NQ * EMBED * 2);
  u16*   h1_bf    = (u16*)  carve((size_t)NQ * FFN * 2);
  u16*   wt_off   = (u16*)  carve((size_t)EMBED * NOFF * 2);   // transposed (N x K)
  u16*   wt_att   = (u16*)  carve((size_t)EMBED * NATT * 2);
  u16*   wt_val   = (u16*)  carve((size_t)EMBED * EMBED * 2);
  u16*   wt_out   = (u16*)  carve((size_t)EMBED * EMBED * 2);
  u16*   wt_1     = (u16*)  carve((size_t)EMBED * FFN * 2);
  u16*   wt_2     = (u16*)  carve((size_t)FFN * EMBED * 2);

  auto cvt = [&](const float* s, u16* d, int n) {
    cvt_bf16_kernel<<<(n + 255) / 256, 256, 0, stream>>>(s, d, n);
  };
  auto cvt_t = [&](const float* s, u16* d, int K, int N) {
    cvt_bf16_t_kernel<<<(K * N + 255) / 256, 256, 0, stream>>>(s, d, K, N);
  };
  auto gemm = [&](const u16* A, const u16* BT, const float* bias, const float* a0,
                  const float* a1, void* C, int M, int N, int K, int epi) {
    dim3 g((M + 255) / 256, N / 32);
    gemm_bf16_wmma<<<g, 256, 0, stream>>>(A, BT, bias, a0, a1, C, M, N, K, epi);
  };

  // 0) precision conversions; weights transposed to N-major for direct B fragments
  cvt(value, value_bf, NV * EMBED);
  cvt_t(Woff, wt_off, EMBED, NOFF);
  cvt_t(Watt, wt_att, EMBED, NATT);
  cvt_t(Wval, wt_val, EMBED, EMBED);
  cvt_t(Wout, wt_out, EMBED, EMBED);
  cvt_t(W1,   wt_1,   EMBED, FFN);
  cvt_t(W2,   wt_2,   FFN, EMBED);

  // 1) LN1 -> identity (f32) and qp = LN1(query)+query_pos (bf16)
  ln_kernel<<<NQ, 256, 0, stream>>>(query, qpos, n1g, n1b, q_id, qp_bf);

  // 2) projection GEMMs (WMMA bf16)
  gemm(qp_bf, wt_off, boff, nullptr, nullptr, offs_f, NQ, NOFF, EMBED, EPI_F32);
  gemm(qp_bf, wt_att, batt, nullptr, nullptr, aw_f, NQ, NATT, EMBED, EPI_F32);
  gemm(value_bf, wt_val, bval, nullptr, nullptr, v_bf, NV, EMBED, EMBED, EPI_BF16);

  // 3) softmax over 12 sampling points per (q, head)
  softmax12_kernel<<<(NQ * HEADS + 255) / 256, 256, 0, stream>>>(aw_f, NQ * HEADS);

  // 4) multi-scale deformable bilinear sampling (wave per (q,head))
  sample_kernel<<<NQ, 256, 0, stream>>>(v_bf, offs_f, aw_f, refp, samp_bf);

  // 5) output projection with fused residuals: x = samp@Wout + bout + identity + query
  gemm(samp_bf, wt_out, bout, q_id, query, x_f, NQ, EMBED, EMBED, EPI_F32);

  // 6) LN2 -> h (bf16)
  ln_kernel<<<NQ, 256, 0, stream>>>(x_f, nullptr, n2g, n2b, nullptr, h_bf);

  // 7) FFN: h1 = gelu(h@W1+b1) (bf16); out = h1@W2 + b2 + x
  gemm(h_bf, wt_1, b1, nullptr, nullptr, h1_bf, NQ, FFN, EMBED, EPI_GELU_BF16);
  gemm(h1_bf, wt_2, b2, x_f, nullptr, d_out, NQ, EMBED, FFN, EPI_F32);
}
